// DeformableConv2D_81810537054370
// MI455X (gfx1250) — compile-verified, matrix operand order checked by reading, not run on hardware
//
#include <hip/hip_runtime.h>
#include <hip/hip_bf16.h>

// ---------------------------------------------------------------------------
// Deformable conv (G=2 groups, K=3) for MI455X / gfx1250 (wave32, WMMA).
// Pipeline per group: offset-conv -> bilinear sample (bf16) -> depthwise 3x3
// (bf16) -> pointwise GEMM 65536x576 @ 576x64 via v_wmma_f32_16x16x32_bf16.
// ---------------------------------------------------------------------------

#define BB 4
#define HH 128
#define WW 128
#define CC 128
#define GG 2
#define CG 64
#define KK 3
#define KIN 576   // 9 * 64
#define FG 64
#define PIX (BB*HH*WW)   // 65536

typedef __attribute__((ext_vector_type(16))) __bf16 v16bf;
typedef __attribute__((ext_vector_type(8)))  float  v8f;

__device__ __forceinline__ unsigned short f2bf(float f) {
    unsigned int u = __float_as_uint(f);
    unsigned int r = u + 0x7FFFu + ((u >> 16) & 1u);   // round-to-nearest-even
    return (unsigned short)(r >> 16);
}
__device__ __forceinline__ float bf2f(unsigned short h) {
    return __uint_as_float(((unsigned int)h) << 16);
}

// ---------------------------------------------------------------------------
// Kernel 1: offset conv. 3x3, Cg=64 -> 18. One thread per pixel.
// Weights (9*64*18 fp32 = 41472 B) staged in LDS, broadcast reads.
// ---------------------------------------------------------------------------
__global__ void k_offsets(const float* __restrict__ x,
                          const float* __restrict__ offw,
                          const float* __restrict__ offb,
                          float* __restrict__ off_out, int g) {
    __shared__ float wsm[9 * CG * 18];
    for (int i = threadIdx.x; i < 9 * CG * 18; i += 256)
        wsm[i] = offw[g * 9 * CG * 18 + i];
    __syncthreads();

    int pixel = blockIdx.x * 256 + threadIdx.x;
    int b = pixel >> 14;
    int rem = pixel & 16383;
    int h = rem >> 7;
    int w = rem & 127;

    float acc[18];
#pragma unroll
    for (int j = 0; j < 18; ++j) acc[j] = offb[g * 18 + j];

    for (int ky = 0; ky < 3; ++ky) {
        int hh = h + ky - 1;
        if (hh < 0 || hh >= HH) continue;
        for (int kx = 0; kx < 3; ++kx) {
            int ww = w + kx - 1;
            if (ww < 0 || ww >= WW) continue;
            const float* xp = x + ((size_t)(b * HH + hh) * WW + ww) * CC + g * CG;
            const float* wp = &wsm[(ky * 3 + kx) * CG * 18];
            for (int c = 0; c < CG; ++c) {
                float xv = xp[c];
                const float* wc = wp + c * 18;
#pragma unroll
                for (int j = 0; j < 18; ++j) acc[j] += xv * wc[j];
            }
        }
    }
    float* op = off_out + (size_t)pixel * 18;
#pragma unroll
    for (int j = 0; j < 18; ++j) op[j] = acc[j];
}

// ---------------------------------------------------------------------------
// Kernel 2: bilinear sampling. One thread per (pixel, tap); 64 channels via
// float4 gathers (channels contiguous in NHWC). Output bf16 into s1[pix][576].
// ---------------------------------------------------------------------------
__global__ void k_sample(const float* __restrict__ x,
                         const float* __restrict__ off_buf,
                         unsigned short* __restrict__ s1, int g) {
    int tid = blockIdx.x * 256 + threadIdx.x;
    if (tid >= PIX * 9) return;
    int pixel = tid / 9;
    int k = tid - pixel * 9;

    int b = pixel >> 14;
    int rem = pixel & 16383;
    int h = rem >> 7;
    int w = rem & 127;

    float dx = (float)(k % 3) - 1.0f;
    float dy = (float)(k / 3) - 1.0f;
    float ox = off_buf[(size_t)pixel * 18 + 2 * k];
    float oy = off_buf[(size_t)pixel * 18 + 2 * k + 1];

    float fx = fminf(fmaxf((float)w + dx + ox, 0.0f), (float)(WW - 1));
    float fy = fminf(fmaxf((float)h + dy + oy, 0.0f), (float)(HH - 1));

    float x0 = floorf(fx), y0 = floorf(fy);
    float x1 = fminf(fmaxf(x0 + 1.0f, 0.0f), (float)(WW - 1));
    x0 = fminf(fmaxf(x0, 0.0f), (float)(WW - 1));
    float y1 = fminf(fmaxf(y0 + 1.0f, 0.0f), (float)(HH - 1));
    y0 = fminf(fmaxf(y0, 0.0f), (float)(HH - 1));

    float wa = (x1 - fx) * (y1 - fy);
    float wb = (x1 - fx) * (fy - y0);
    float wc = (fx - x0) * (y1 - fy);
    float wd = (fx - x0) * (fy - y0);

    int x0i = (int)x0, x1i = (int)x1, y0i = (int)y0, y1i = (int)y1;
    const float* pa = x + ((size_t)((b * HH + y0i) * WW + x0i)) * CC + g * CG;
    const float* pb = x + ((size_t)((b * HH + y1i) * WW + x0i)) * CC + g * CG;
    const float* pc = x + ((size_t)((b * HH + y0i) * WW + x1i)) * CC + g * CG;
    const float* pd = x + ((size_t)((b * HH + y1i) * WW + x1i)) * CC + g * CG;

    unsigned short* op = s1 + (size_t)pixel * KIN + k * CG;
    for (int c = 0; c < CG; c += 4) {
        float4 A = *(const float4*)(pa + c);
        float4 Bv = *(const float4*)(pb + c);
        float4 Cv = *(const float4*)(pc + c);
        float4 Dv = *(const float4*)(pd + c);
        float r0 = wa * A.x + wb * Bv.x + wc * Cv.x + wd * Dv.x;
        float r1 = wa * A.y + wb * Bv.y + wc * Cv.y + wd * Dv.y;
        float r2 = wa * A.z + wb * Bv.z + wc * Cv.z + wd * Dv.z;
        float r3 = wa * A.w + wb * Bv.w + wc * Cv.w + wd * Dv.w;
        uint2 packed;
        packed.x = (unsigned)f2bf(r0) | ((unsigned)f2bf(r1) << 16);
        packed.y = (unsigned)f2bf(r2) | ((unsigned)f2bf(r3) << 16);
        *(uint2*)(op + c) = packed;
    }
}

// ---------------------------------------------------------------------------
// Kernel 3: depthwise 3x3 over 576 channels. One thread per (pixel, 8-ch).
// bf16 in / fp32 accumulate / bf16 out.
// ---------------------------------------------------------------------------
__global__ void k_depthwise(const unsigned short* __restrict__ s1,
                            const float* __restrict__ dww,
                            const float* __restrict__ dwb,
                            unsigned short* __restrict__ s2, int g) {
    int tid = blockIdx.x * 256 + threadIdx.x;
    if (tid >= PIX * (KIN / 8)) return;
    int pixel = tid / (KIN / 8);
    int chunk = tid - pixel * (KIN / 8);
    int c0 = chunk * 8;

    int b = pixel >> 14;
    int rem = pixel & 16383;
    int h = rem >> 7;
    int w = rem & 127;

    float acc[8];
#pragma unroll
    for (int i = 0; i < 8; ++i) acc[i] = dwb[g * KIN + c0 + i];

    for (int ky = 0; ky < 3; ++ky) {
        int hh = h + ky - 1;
        if (hh < 0 || hh >= HH) continue;
        for (int kx = 0; kx < 3; ++kx) {
            int ww = w + kx - 1;
            if (ww < 0 || ww >= WW) continue;
            size_t npix = (size_t)b * (HH * WW) + (size_t)hh * WW + ww;
            uint4 v = *(const uint4*)(s1 + npix * KIN + c0);
            const unsigned short* hp = (const unsigned short*)&v;
            const float* wt = dww + (size_t)g * 9 * KIN + (ky * 3 + kx) * KIN + c0;
#pragma unroll
            for (int i = 0; i < 8; ++i) acc[i] += bf2f(hp[i]) * wt[i];
        }
    }
    unsigned short hv[8];
#pragma unroll
    for (int i = 0; i < 8; ++i) hv[i] = f2bf(acc[i]);
    *(uint4*)(s2 + (size_t)pixel * KIN + c0) = *(const uint4*)hv;
}

// ---------------------------------------------------------------------------
// Kernel 4: pointwise GEMM  [65536 x 576](bf16) x [576 x 64](bf16->LDS) + bias
// via v_wmma_f32_16x16x32_bf16. Block = 256 thr = 8 wave32; each wave owns a
// 16-row x 64-col strip (4 accumulators); 18 K-steps of 32.
// B staged transposed [N][K] in LDS with padded stride 584 (16B aligned,
// conflict-free ds_load_b128 across the 16 N-lanes).
// ---------------------------------------------------------------------------
#define BSTRIDE 584

__global__ void k_pointwise(const unsigned short* __restrict__ s2,
                            const float* __restrict__ pww,
                            const float* __restrict__ pwb,
                            float* __restrict__ out, int g) {
    __shared__ unsigned short Blds[FG * BSTRIDE];   // 74752 B

    for (int idx = threadIdx.x; idx < KIN * FG; idx += 256) {
        int k = idx >> 6;          // K index
        int n = idx & 63;          // N index
        Blds[n * BSTRIDE + k] = f2bf(pww[(size_t)g * KIN * FG + idx]);
    }
    __syncthreads();

    int wave = threadIdx.x >> 5;
    int lane = threadIdx.x & 31;
    int m    = lane & 15;          // row within tile / column within N-tile
    int half = lane >> 4;          // K-half selector per ISA 16-bit layout

    int rowBase = blockIdx.x * 128 + wave * 16;
    const unsigned short* aRow = s2 + (size_t)(rowBase + m) * KIN;

    v8f acc[4] = {};

    union Frag { v16bf v; uint4 q[2]; };

    for (int k0 = 0; k0 < KIN; k0 += 32) {
        if (k0 + 64 < KIN)
            __builtin_prefetch(aRow + k0 + 64, 0, 0);   // global_prefetch_b8

        Frag af;
        // lane half 0: K = k0+0..7 (elems 0-7), k0+16..23 (elems 8-15)
        // lane half 1: K = k0+8..15,            k0+24..31
        af.q[0] = *(const uint4*)(aRow + k0 + half * 8);
        af.q[1] = *(const uint4*)(aRow + k0 + 16 + half * 8);

#pragma unroll
        for (int a = 0; a < 4; ++a) {
            const unsigned short* bp = &Blds[(a * 16 + m) * BSTRIDE + k0 + half * 8];
            Frag bf;
            bf.q[0] = *(const uint4*)bp;          // K k0.. (half-selected)
            bf.q[1] = *(const uint4*)(bp + 16);   // K k0+16..
            acc[a] = __builtin_amdgcn_wmma_f32_16x16x32_bf16(
                false, af.v, false, bf.v, (short)0, acc[a], false, false);
        }
    }

    // Epilogue: D layout -> VGPR r holds M=r (lanes 0-15) / M=r+8 (lanes 16-31),
    // N = lane & 15 within each 16-col tile.
#pragma unroll
    for (int a = 0; a < 4; ++a) {
        int col = a * 16 + m;
        float bias = pwb[g * FG + col];
#pragma unroll
        for (int r = 0; r < 8; ++r) {
            int row = rowBase + r + 8 * half;
            out[(size_t)row * (GG * FG) + g * FG + col] = acc[a][r] + bias;
        }
    }
}

// ---------------------------------------------------------------------------
extern "C" void kernel_launch(void* const* d_in, const int* in_sizes, int n_in,
                              void* d_out, int out_size, void* d_ws, size_t ws_size,
                              hipStream_t stream) {
    (void)in_sizes; (void)n_in; (void)out_size; (void)ws_size;
    const float* x    = (const float*)d_in[0];
    const float* offw = (const float*)d_in[1];
    const float* offb = (const float*)d_in[2];
    const float* dww  = (const float*)d_in[3];
    const float* dwb  = (const float*)d_in[4];
    const float* pww  = (const float*)d_in[5];
    const float* pwb  = (const float*)d_in[6];
    float* out = (float*)d_out;

    char* ws = (char*)d_ws;
    float* off_buf          = (float*)ws;                                   // 4.72 MB
    unsigned short* s1      = (unsigned short*)(ws + (size_t)PIX * 18 * 4); // 75.5 MB
    unsigned short* s2      = s1 + (size_t)PIX * KIN;                       // 75.5 MB

    for (int g = 0; g < GG; ++g) {
        k_offsets  <<<PIX / 256,            256, 0, stream>>>(x, offw, offb, off_buf, g);
        k_sample   <<<(PIX * 9) / 256,      256, 0, stream>>>(x, off_buf, s1, g);
        k_depthwise<<<(PIX * (KIN/8))/256,  256, 0, stream>>>(s1, dww, dwb, s2, g);
        k_pointwise<<<PIX / 128,            256, 0, stream>>>(s2, pww, pwb, out, g);
    }
}
